// _FlashDiffCustomAttention_34222299414954
// MI455X (gfx1250) — matrix operand order repaired and checked by reading, not compiled, not run on hardware
//
#include <hip/hip_runtime.h>
#include <stdint.h>

typedef __attribute__((ext_vector_type(16))) _Float16 v16h;
typedef __attribute__((ext_vector_type(8)))  float    v8f;
typedef __attribute__((ext_vector_type(8)))  uint32_t v8u;

#define B_   2
#define S_   2048
#define H_   8      // score heads
#define H2_  16     // stored head-pairs
#define D_   128
#define E_   256    // 2*D concat value dim
#define BM   64     // query rows per workgroup
#define BN   32     // keys per iteration
#define EH   128    // value columns per wave (one half of E)
#define NWAVE 16
#define NTHREADS 512

#define KPAD (D_ + 8)   // padded K row (halves)
#define VPAD (BN + 8)   // padded VT/P row (halves)

#define LAMBDA_INIT_F 0.78360576653162444f
#define OUT_SCALE_F   0.21639423346837556f   // 1 - lambda_init
#define NEG_INF_F     (-1e30f)
#define QK_SCALE_F    0.08838834764831843f   // 1/sqrt(128), folded into Q fragments
#define EPS_F         1e-5f

#define KH_ELEMS ((size_t)B_ * H2_ * S_ * D_)   // 8388608

// Even K index of the (v, s=0) half-pair for 16-bit A fragments (ISA 16x32 A layout)
__device__ __forceinline__ int a_pat(int v, int half) {
  return ((v & 4) << 2) + ((v & 3) << 1) + (half << 3);
}
// Even K index for 16-bit B fragments (32x16): same K order as a_pat
__device__ __forceinline__ int b_pat(int v, int half) {
  return ((v & 4) << 2) + ((v & 3) << 1) + (half << 3);
}

// ---- pre-pass: fp32 -> fp16 K/V relayout (once; tables then live in L2) ----
//   kh[b][h2][s][d]          (row-major K, fp16)
//   vt[b][h][e(256)][s]      (transposed concat V, fp16)
__global__ __launch_bounds__(256)
void convert_kv_kernel(const float* __restrict__ k, const float* __restrict__ v,
                       _Float16* __restrict__ kh, _Float16* __restrict__ vt) {
  const size_t i = (size_t)blockIdx.x * 256 + threadIdx.x;   // < 8388608
  {
    const int d  = (int)(i & 127);
    const int s  = (int)((i >> 7) & 2047);
    const int h2 = (int)((i >> 18) & 15);
    const int b  = (int)(i >> 22);
    kh[i] = (_Float16)k[(((size_t)b * S_ + s) * H2_ + h2) * D_ + d];
  }
  {
    const int s  = (int)(i & 2047);
    const int e  = (int)((i >> 11) & 255);
    const int hh = (int)((i >> 19) & 7);
    const int b  = (int)(i >> 22);
    vt[i] = (_Float16)v[(((size_t)b * S_ + s) * H2_ + (2 * hh + (e >> 7))) * D_ + (e & 127)];
  }
}

__global__ __launch_bounds__(NTHREADS)
void diff_flash_attn_kernel(const float* __restrict__ q,
                            const _Float16* __restrict__ kh, const _Float16* __restrict__ vt,
                            const float* __restrict__ lq1, const float* __restrict__ lk1,
                            const float* __restrict__ lq2, const float* __restrict__ lk2,
                            const float* __restrict__ w, float* __restrict__ out) {
  // Double-buffered async staging overlaid with the finalize staging buffer.
  union __align__(16) Smem {
    struct {
      _Float16 sK[2][2][BN][KPAD];    // [buf][branch][key][d]
      _Float16 sVT[2][E_][VPAD];      // [buf][e][key]
    } ld;
    float sO[4][2][16][EH];           // [wm][eh][row][col] lam*o2 staging (post-loop)
  };
  __shared__ Smem su;
  __shared__ _Float16 sP[NWAVE][16][VPAD];  // per-wave P relayout scratch
  __shared__ float    sStat[4][2][16];

  const int tid  = threadIdx.x;
  const int lane = tid & 31;
  const int wave = tid >> 5;
  const int wm   = wave & 3;          // which 16-row query tile
  const int j    = wave >> 2;         // 0..3
  const int a    = j & 1;             // attention branch handled by this wave
  const int eh   = j >> 1;            // which 128-wide value half
  const int half = lane >> 4;
  const int ln   = lane & 15;

  const int qb = blockIdx.x;
  const int h  = blockIdx.y;
  const int b  = blockIdx.z;

  const int qrow0 = qb * BM + wm * 16;

  // lambda_full (redundant per-thread, tiny)
  float l1 = 0.f, l2 = 0.f;
  for (int i = 0; i < D_; ++i) { l1 += lq1[i] * lk1[i]; l2 += lq2[i] * lk2[i]; }
  const float lam = __expf(l1) - __expf(l2) + LAMBDA_INIT_F;

  // ---- async DMA issue: 4 x global_load_async_to_lds_b128 per wave per tile ----
  const int kb  = wave >> 3;          // K branch this wave copies
  const int kr0 = (wave & 7) * 4;     // 4 K rows per wave
  const int ve0 = wave * 16;          // 16 VT rows per wave
  const uint32_t kvo_lane = (uint32_t)((lane >> 4) * (D_ * 2) + (lane & 15) * 16);
  const uint32_t vvo_lane = (uint32_t)((lane >> 2) * (S_ * 2) + (lane & 3) * 16);

  auto issue_tile = [&](int n0, int buf) {
    const _Float16* ksrc = kh + (((size_t)(b * H2_ + 2 * h + kb) * S_) + n0) * D_ + (size_t)kr0 * D_;
    const _Float16* vsrc = vt + ((size_t)(b * H_ + h) * E_ + ve0) * S_ + n0;
    const uint32_t klds = (uint32_t)(uintptr_t)(&su.ld.sK[buf][kb][kr0][0])
                          + (uint32_t)((lane >> 4) * (KPAD * 2)) + (uint32_t)((lane & 15) * 16);
    const uint32_t vlds = (uint32_t)(uintptr_t)(&su.ld.sVT[buf][ve0][0])
                          + (uint32_t)((lane >> 2) * (VPAD * 2)) + (uint32_t)((lane & 3) * 16);
    asm volatile("global_load_async_to_lds_b128 %0, %1, %2"
                 :: "v"(klds), "v"(kvo_lane), "s"(ksrc) : "memory");
    asm volatile("global_load_async_to_lds_b128 %0, %1, %2"
                 :: "v"(klds + (uint32_t)(2 * KPAD * 2)), "v"(kvo_lane + (uint32_t)(2 * D_ * 2)),
                    "s"(ksrc) : "memory");
    asm volatile("global_load_async_to_lds_b128 %0, %1, %2"
                 :: "v"(vlds), "v"(vvo_lane), "s"(vsrc) : "memory");
    asm volatile("global_load_async_to_lds_b128 %0, %1, %2"
                 :: "v"(vlds + (uint32_t)(8 * VPAD * 2)), "v"(vvo_lane + (uint32_t)(8 * S_ * 2)),
                    "s"(vsrc) : "memory");
  };

  // ---- Q fragments for this wave's branch (A layout, 1/sqrt(D) folded in) ----
  v16h Aq[4];
  {
    const size_t base = (((size_t)b * S_ + (qrow0 + ln)) * H2_ + (2 * h + a)) * D_;
    for (int c = 0; c < 4; ++c) {
      v8u pk;
      for (int r = 0; r < 8; ++r) {
        const int d0 = c * 32 + a_pat(r, half);
        const float2 f = *(const float2*)(q + base + d0);
        const uint16_t ux = __builtin_bit_cast(uint16_t, (_Float16)(f.x * QK_SCALE_F));
        const uint16_t uy = __builtin_bit_cast(uint16_t, (_Float16)(f.y * QK_SCALE_F));
        pk[r] = (uint32_t)ux | ((uint32_t)uy << 16);
      }
      Aq[c] = __builtin_bit_cast(v16h, pk);
    }
  }

  const v8f vzero = {0.f, 0.f, 0.f, 0.f, 0.f, 0.f, 0.f, 0.f};
  v8f accO[8];
  float mrun[8], lrun[8];
  for (int t = 0; t < 8; ++t) accO[t] = vzero;
  for (int r = 0; r < 8; ++r) { mrun[r] = NEG_INF_F; lrun[r] = 0.f; }

  const int nblocks = 2 * (qb + 1);  // causal; always >= 2
  issue_tile(0, 0);
  issue_tile(BN, 1);

  for (int nb = 0; nb < nblocks; ++nb) {
    const int n0 = nb * BN;
    const int buf = nb & 1;

    // drain this tile's 4 async copies (leave the prefetched tile in flight)
    if (nb + 1 < nblocks) asm volatile("s_wait_asynccnt 4" ::: "memory");
    else                  asm volatile("s_wait_asynccnt 0" ::: "memory");
    __syncthreads();

    // ---- QK: preload ALL B fragments, then stream 8 WMMAs back-to-back ----
    v8f c0 = vzero, c1 = vzero;
    {
      const uint32_t* kr0p = (const uint32_t*)&su.ld.sK[buf][a][ln][0];
      const uint32_t* kr1p = (const uint32_t*)&su.ld.sK[buf][a][16 + ln][0];
      v8u bq[8];
      for (int kc = 0; kc < 4; ++kc) {
        for (int r = 0; r < 8; ++r) {
          const int idx = (kc * 32 + b_pat(r, half)) >> 1;
          bq[kc * 2 + 0][r] = kr0p[idx];
          bq[kc * 2 + 1][r] = kr1p[idx];
        }
      }
      for (int kc = 0; kc < 4; ++kc) {
        c0 = __builtin_amdgcn_wmma_f32_16x16x32_f16(false, Aq[kc], false,
                 __builtin_bit_cast(v16h, bq[kc * 2 + 0]), (short)0, c0, false, false);
        c1 = __builtin_amdgcn_wmma_f32_16x16x32_f16(false, Aq[kc], false,
                 __builtin_bit_cast(v16h, bq[kc * 2 + 1]), (short)0, c1, false, false);
      }
    }

    // ---- causal mask: only diagonal-touching blocks (wave-uniform branch) ----
    if (n0 + BN - 1 > qrow0) {
      const int kg0 = n0 + ln, kg1 = n0 + 16 + ln;
      for (int r = 0; r < 8; ++r) {
        const int qg = qrow0 + r + 8 * half;
        if (kg0 > qg) c0[r] = NEG_INF_F;
        if (kg1 > qg) c1[r] = NEG_INF_F;
      }
    }

    // ---- online softmax (fp32) ----
    float p0[8], p1[8], alpha[8];
    for (int r = 0; r < 8; ++r) {
      float m = fmaxf(c0[r], c1[r]);
      for (int msk = 1; msk < 16; msk <<= 1) m = fmaxf(m, __shfl_xor(m, msk, 32));
      const float mnew = fmaxf(mrun[r], m);
      alpha[r] = __expf(mrun[r] - mnew);
      p0[r] = __expf(c0[r] - mnew);
      p1[r] = __expf(c1[r] - mnew);
      float rs = p0[r] + p1[r];
      for (int msk = 1; msk < 16; msk <<= 1) rs += __shfl_xor(rs, msk, 32);
      lrun[r] = lrun[r] * alpha[r] + rs;
      mrun[r] = mnew;
    }

    // ---- P: D layout -> A layout via wave-private LDS ----
    for (int r = 0; r < 8; ++r) {
      const int m = r + 8 * half;
      sP[wave][m][ln]      = (_Float16)p0[r];
      sP[wave][m][16 + ln] = (_Float16)p1[r];
    }

    // rescale accumulators while the P stores land
    for (int t = 0; t < 8; ++t)
      for (int r = 0; r < 8; ++r)
        accO[t][r] *= alpha[r];

    v8u pu;
    const uint32_t* prow = (const uint32_t*)&sP[wave][ln][0];
    for (int r = 0; r < 8; ++r) pu[r] = prow[a_pat(r, half) >> 1];
    const v16h Pa = __builtin_bit_cast(v16h, pu);

    // ---- PV: preload ALL B fragments, then stream 8 WMMAs ----
    {
      v8u bv[8];
      for (int t = 0; t < 8; ++t) {
        const int e = eh * EH + t * 16 + ln;
        const uint32_t* vr = (const uint32_t*)&su.ld.sVT[buf][e][0];
        for (int r = 0; r < 8; ++r) bv[t][r] = vr[b_pat(r, half) >> 1];
      }
      for (int t = 0; t < 8; ++t)
        accO[t] = __builtin_amdgcn_wmma_f32_16x16x32_f16(false, Pa, false,
                      __builtin_bit_cast(v16h, bv[t]), (short)0, accO[t], false, false);
    }
    __syncthreads();   // everyone done reading buf before it is refilled

    if (nb + 2 < nblocks) issue_tile(n0 + 2 * BN, buf);
  }

  // ---- finalize (all async traffic drained; union region reused as sO) ----
  float invl[8];
  for (int r = 0; r < 8; ++r) invl[r] = 1.0f / lrun[r];

  if (a == 1) {
    for (int t = 0; t < 8; ++t)
      for (int r = 0; r < 8; ++r)
        su.sO[wm][eh][r + 8 * half][t * 16 + ln] = lam * accO[t][r] * invl[r];
  }
  __syncthreads();

  float res[8][8];
  if (a == 0) {
    for (int t = 0; t < 8; ++t)
      for (int r = 0; r < 8; ++r)
        res[t][r] = accO[t][r] * invl[r] - su.sO[wm][eh][r + 8 * half][t * 16 + ln];
    for (int r = 0; r < 8; ++r) {
      float ss = 0.f;
      for (int t = 0; t < 8; ++t) ss += res[t][r] * res[t][r];
      for (int msk = 1; msk < 16; msk <<= 1) ss += __shfl_xor(ss, msk, 32);
      if (ln == 0) sStat[wm][eh][r + 8 * half] = ss;
    }
  }
  __syncthreads();

  if (a == 0) {
    for (int r = 0; r < 8; ++r) {
      const int m = r + 8 * half;
      const float tot = sStat[wm][0][m] + sStat[wm][1][m];
      const float rms = rsqrtf(tot * (1.0f / (float)E_) + EPS_F);
      const int srow = qrow0 + m;
      for (int t = 0; t < 8; ++t) {
        const int e = eh * EH + t * 16 + ln;
        out[(((size_t)b * S_ + srow) * H2_ + (2 * h + eh)) * D_ + (e & 127)] =
            w[e] * res[t][r] * rms * OUT_SCALE_F;
      }
    }
  }
}

extern "C" void kernel_launch(void* const* d_in, const int* in_sizes, int n_in,
                              void* d_out, int out_size, void* d_ws, size_t ws_size,
                              hipStream_t stream) {
  (void)in_sizes; (void)n_in; (void)out_size; (void)ws_size;
  const float* q   = (const float*)d_in[0];
  const float* k   = (const float*)d_in[1];
  const float* v   = (const float*)d_in[2];
  const float* lq1 = (const float*)d_in[3];
  const float* lk1 = (const float*)d_in[4];
  const float* lq2 = (const float*)d_in[5];
  const float* lk2 = (const float*)d_in[6];
  const float* w   = (const float*)d_in[7];
  float* out = (float*)d_out;

  _Float16* kh = (_Float16*)d_ws;             // 16.78 MB
  _Float16* vt = kh + KH_ELEMS;               // 16.78 MB

  convert_kv_kernel<<<(int)(KH_ELEMS / 256), 256, 0, stream>>>(k, v, kh, vt);

  dim3 grid(S_ / BM, H_, B_);
  diff_flash_attn_kernel<<<grid, NTHREADS, 0, stream>>>(q, kh, vt, lq1, lk1, lq2, lk2, w, out);
}
